// Encoder_HieStackedCorr_37211596653081
// MI455X (gfx1250) — matrix operand order repaired and checked by reading, not compiled, not run on hardware
//
#include <hip/hip_runtime.h>
#include <hip/hip_bf16.h>

// ---------------------------------------------------------------------------
// Encoder_HieStackedCorr for MI455X (gfx1250): bf16 WMMA GEMMs + fused
// mean-over-n algebraic reduction (never materializes normalized uncorr).
// Round 2: async global->LDS staging (ASYNCcnt), double-buffered GEMM tiles,
// native bf16 converts, coalesced epilogue stores.
// ---------------------------------------------------------------------------

typedef __attribute__((ext_vector_type(16))) __bf16         v16bf;
typedef __attribute__((ext_vector_type(8)))  float          v8f;
typedef __attribute__((ext_vector_type(16))) unsigned short v16u;
typedef __attribute__((ext_vector_type(8)))  unsigned short v8u;

#define B_     64
#define N_     256
#define VDIM_  2048
#define LR_    64
#define EMBED_ 1024

__device__ __forceinline__ unsigned short f2bf(float f) {
  return __builtin_bit_cast(unsigned short, (__bf16)f);   // native v_cvt, RNE
}
__device__ __forceinline__ float bf2f(unsigned short u) {
  return (float)__builtin_bit_cast(__bf16, u);
}
__device__ __forceinline__ v8f zero8() {
  v8f z;
#pragma unroll
  for (int i = 0; i < 8; i++) z[i] = 0.f;
  return z;
}

// async global->LDS copy of 16 bytes per lane (gfx1250 ASYNCcnt path)
__device__ __forceinline__ void async_copy_b128(unsigned lds_addr,
                                                const void* gaddr) {
  asm volatile("global_load_async_to_lds_b128 %0, %1, off"
               :
               : "v"(lds_addr), "v"((unsigned long long)(size_t)gaddr)
               : "memory");
}
__device__ __forceinline__ void wait_async0() {
  asm volatile("s_wait_asynccnt 0x0" ::: "memory");
}

// A fragment (16x32 bf16, M x K), LDS row-major [rows][ldk].
// Lane lo = row; K elements = {8*hi .. 8*hi+7} U {16+8*hi .. 16+8*hi+7}.
__device__ __forceinline__ v16bf load_a_frag(const unsigned short* lds,
                                             int row_base, int ldk, int kbase,
                                             int lane) {
  int lo = lane & 15, hi = lane >> 4;
  const unsigned short* p = lds + (size_t)(row_base + lo) * ldk + kbase;
  v8u a0 = *(const v8u*)(p + 8 * hi);
  v8u a1 = *(const v8u*)(p + 16 + 8 * hi);
  v16u r;
#pragma unroll
  for (int i = 0; i < 8; i++) { r[i] = a0[i]; r[i + 8] = a1[i]; }
  return __builtin_bit_cast(v16bf, r);
}

// B fragment (32x16 bf16, K x N), LDS stored n-major [ncols][ldk].
// Lane lo = column; K elements = {8*hi .. 8*hi+7} U {16+8*hi .. 16+8*hi+7}.
__device__ __forceinline__ v16bf load_b_frag(const unsigned short* lds,
                                             int col_base, int ldk, int kbase,
                                             int lane) {
  int lo = lane & 15, hi = lane >> 4;
  const unsigned short* p = lds + (size_t)(col_base + lo) * ldk + kbase;
  v8u b0 = *(const v8u*)(p + 8 * hi);
  v8u b1 = *(const v8u*)(p + 16 + 8 * hi);
  v16u r;
#pragma unroll
  for (int i = 0; i < 8; i++) { r[i] = b0[i]; r[i + 8] = b1[i]; }
  return __builtin_bit_cast(v16bf, r);
}

__device__ __forceinline__ v8f wmma_bf16(v16bf a, v16bf b, v8f c) {
  return __builtin_amdgcn_wmma_f32_16x16x32_bf16(false, a, false, b,
                                                 (short)0, c, false, false);
}

// ---------------------------------------------------------------------------
// K1: weight-norm scales  scale[k] = g[k] * rsqrt(sum_v v[k,v]^2), k in [0,128)
// ---------------------------------------------------------------------------
__global__ void k_scales(const float* __restrict__ U1_v,
                         const float* __restrict__ U1_g,
                         const float* __restrict__ U2_v,
                         const float* __restrict__ U2_g,
                         float* __restrict__ scale) {
  __shared__ float red[256];
  int r = blockIdx.x;           // 0..127
  int tid = threadIdx.x;
  const float* v = (r < LR_) ? (U1_v + (size_t)r * VDIM_)
                             : (U2_v + (size_t)(r - LR_) * VDIM_);
  float ss = 0.f;
#pragma unroll
  for (int j = 0; j < 8; j++) {
    float x = v[tid + 256 * j];
    ss += x * x;
  }
  red[tid] = ss;
  __syncthreads();
  for (int s = 128; s > 0; s >>= 1) {
    if (tid < s) red[tid] += red[tid + s];
    __syncthreads();
  }
  if (tid == 0) {
    float g = (r < LR_) ? U1_g[r] : U2_g[r - LR_];
    scale[r] = g * rsqrtf(red[0]);
  }
}

// ---------------------------------------------------------------------------
// K2: fused dual projection.  P[b][n][k2] = relu(scale[k2]*(Vmat[b,n,:].U[k2,:])
//     + bias[k2]) as bf16. k2<64: right (U1), k2>=64: left (U2).
// Double-buffered LDS tiles; coalesced bf16 output via LDS staging.
// Grid (4, 64): mblock, batch.  256 threads = 8 waves.
// ---------------------------------------------------------------------------
__global__ void k_proj(const float* __restrict__ Vmat,
                       const float* __restrict__ U1_v,
                       const float* __restrict__ U2_v,
                       const float* __restrict__ U1_b,
                       const float* __restrict__ U2_b,
                       const float* __restrict__ scale,
                       unsigned short* __restrict__ P) {
  __shared__ unsigned short Alds[2][64 * 32];   // 4 KB x2
  __shared__ unsigned short Blds[2][128 * 32];  // 8 KB x2
  __shared__ unsigned short Olds[64 * 128];     // 16 KB output tile

  int tid = threadIdx.x;
  int lane = tid & 31, w = tid >> 5;
  int b = blockIdx.y;
  int n0 = blockIdx.x * 64;

  v8f acc[4];
#pragma unroll
  for (int mt = 0; mt < 4; mt++) acc[mt] = zero8();

  int rA = tid >> 2, kA = (tid & 3) * 8;       // A staging: 8 elems/thread
  int rB = tid >> 1, kB = (tid & 1) * 16;      // B staging: 16 elems/thread

  const float* srcBrow = (rB < LR_) ? (U1_v + (size_t)rB * VDIM_ + kB)
                                    : (U2_v + (size_t)(rB - LR_) * VDIM_ + kB);
  const float* srcArow = Vmat + ((size_t)(b * N_ + n0 + rA)) * VDIM_ + kA;

  auto stage = [&](int buf, int kt) {
    int v0 = kt * 32;
    float a[8], bb[16];
#pragma unroll
    for (int j = 0; j < 8; j++) a[j] = srcArow[v0 + j];
#pragma unroll
    for (int j = 0; j < 16; j++) bb[j] = srcBrow[v0 + j];
#pragma unroll
    for (int j = 0; j < 8; j++) Alds[buf][rA * 32 + kA + j] = f2bf(a[j]);
#pragma unroll
    for (int j = 0; j < 16; j++) Blds[buf][rB * 32 + kB + j] = f2bf(bb[j]);
  };

  stage(0, 0);
  int cur = 0;
  for (int kt = 0; kt < VDIM_ / 32; kt++) {
    __syncthreads();
    if (kt + 1 < VDIM_ / 32) stage(cur ^ 1, kt + 1);
    v16bf bf = load_b_frag(Blds[cur], 16 * w, 32, 0, lane);
#pragma unroll
    for (int mt = 0; mt < 4; mt++) {
      v16bf af = load_a_frag(Alds[cur], 16 * mt, 32, 0, lane);
      acc[mt] = wmma_bf16(af, bf, acc[mt]);
    }
    cur ^= 1;
  }

  // epilogue: scale, bias, relu -> LDS tile -> coalesced 16B stores
  int lo = lane & 15, hi = lane >> 4;
  int k2 = 16 * w + lo;
  float sc = scale[k2];
  float bi = (k2 < LR_) ? U1_b[k2] : U2_b[k2 - LR_];
  __syncthreads();
#pragma unroll
  for (int mt = 0; mt < 4; mt++) {
#pragma unroll
    for (int r = 0; r < 8; r++) {
      int row = 16 * mt + r + 8 * hi;
      float v = sc * acc[mt][r] + bi;
      v = v > 0.f ? v : 0.f;
      Olds[row * 128 + k2] = f2bf(v);
    }
  }
  __syncthreads();
#pragma unroll
  for (int c = 0; c < 4; c++) {
    int off = (tid + 256 * c) * 8;    // element offset in 64x128 tile
    int row = off >> 7, col = off & 127;
    *(v8u*)(P + ((size_t)(b * N_ + n0 + row)) * 128 + col) =
        *(const v8u*)(&Olds[off]);
  }
}

// ---------------------------------------------------------------------------
// K3: d[b,i] = rsqrt(dot(left_i, right_i) + 1e-6)
// ---------------------------------------------------------------------------
__global__ void k_diag(const unsigned short* __restrict__ P,
                       float* __restrict__ dbuf) {
  int b = blockIdx.x;
  int i = threadIdx.x;
  const unsigned short* row = P + ((size_t)(b * N_ + i)) * 128;
  float s = 0.f;
#pragma unroll 8
  for (int k = 0; k < LR_; k++) s += bf2f(row[64 + k]) * bf2f(row[k]);
  dbuf[b * N_ + i] = rsqrtf(s + 1e-6f);
}

// ---------------------------------------------------------------------------
// K4: u = left @ right^T with fused reduction:
//   t[m] = sum_n d[n]*u[n,m];  w[b,m] = (N+1 - d[m]*t[m]) / N
// Tiles staged via async global->LDS (no VGPR round-trip; pure bf16 data).
// Grid (4, 64): m-block of 64 columns, batch. 256 threads = 8 waves.
// ---------------------------------------------------------------------------
__global__ void k_colsum(const unsigned short* __restrict__ P,
                         const float* __restrict__ dbuf,
                         float* __restrict__ wbuf) {
  __shared__ unsigned short Alds[256 * 64];  // left: all n x k      (32 KB)
  __shared__ unsigned short Blds[64 * 64];   // right cols, n-major  (8 KB)
  __shared__ float dsh[256];
  __shared__ float tsum[64];

  int tid = threadIdx.x;
  int lane = tid & 31, w = tid >> 5;
  int b = blockIdx.y;
  int m0 = blockIdx.x * 64;

  // stage A = left[b][n][k] = P[b][n][64+k] via async b128 (16B per lane)
#pragma unroll
  for (int j = 0; j < 8; j++) {
    int off = (tid + 256 * j) * 8;   // element offset in 256x64
    int n = off >> 6, kloc = off & 63;
    async_copy_b128((unsigned)(size_t)(&Alds[off]),
                    P + ((size_t)(b * N_ + n)) * 128 + 64 + kloc);
  }
  // stage B = right[b][m0+mloc][k] = P[b][m0+mloc][k]
#pragma unroll
  for (int j = 0; j < 2; j++) {
    int off = (tid + 256 * j) * 8;   // element offset in 64x64
    int mloc = off >> 6, kloc = off & 63;
    async_copy_b128((unsigned)(size_t)(&Blds[off]),
                    P + ((size_t)(b * N_ + m0 + mloc)) * 128 + kloc);
  }
  dsh[tid] = dbuf[b * N_ + tid];
  if (tid < 64) tsum[tid] = 0.f;
  wait_async0();
  __syncthreads();

  // wave w computes rows 32w..32w+31, all 4 column tiles
  v8f acc[2][4];
#pragma unroll
  for (int r2 = 0; r2 < 2; r2++)
#pragma unroll
    for (int ct = 0; ct < 4; ct++) acc[r2][ct] = zero8();

#pragma unroll
  for (int kb = 0; kb < 64; kb += 32) {
    v16bf bf[4];
#pragma unroll
    for (int ct = 0; ct < 4; ct++) bf[ct] = load_b_frag(Blds, 16 * ct, 64, kb, lane);
#pragma unroll
    for (int r2 = 0; r2 < 2; r2++) {
      v16bf af = load_a_frag(Alds, 32 * w + 16 * r2, 64, kb, lane);
#pragma unroll
      for (int ct = 0; ct < 4; ct++) acc[r2][ct] = wmma_bf16(af, bf[ct], acc[r2][ct]);
    }
  }

  // fused d_n-weighted column reduction (LDS float atomics -> ds_add_f32)
  int lo = lane & 15, hi = lane >> 4;
#pragma unroll
  for (int ct = 0; ct < 4; ct++) {
    float part = 0.f;
#pragma unroll
    for (int r2 = 0; r2 < 2; r2++) {
#pragma unroll
      for (int r = 0; r < 8; r++) {
        int n = 32 * w + 16 * r2 + r + 8 * hi;
        part += dsh[n] * acc[r2][ct][r];
      }
    }
    atomicAdd(&tsum[16 * ct + lo], part);
  }
  __syncthreads();

  if (tid < 64) {
    int m = m0 + tid;
    float dm = dsh[m];
    wbuf[b * N_ + m] = (257.0f - dm * tsum[tid]) * (1.0f / 256.0f);
  }
}

// ---------------------------------------------------------------------------
// K5: feats[b,v] = sum_m w[b,m] * Vmat[b,m,v]   (memory bound, float4)
// Grid (2, 64).
// ---------------------------------------------------------------------------
__global__ void k_feats(const float* __restrict__ Vmat,
                        const float* __restrict__ wbuf,
                        float* __restrict__ feats) {
  __shared__ float wsh[256];
  int tid = threadIdx.x;
  int b = blockIdx.y;
  int vbase = blockIdx.x * 1024 + tid * 4;
  wsh[tid] = wbuf[b * N_ + tid];
  __syncthreads();

  float a0 = 0.f, a1 = 0.f, a2 = 0.f, a3 = 0.f;
  const float* base = Vmat + ((size_t)b * N_) * VDIM_ + vbase;
  for (int m = 0; m < N_; m++) {
    if (m + 8 < N_)
      __builtin_prefetch(base + (size_t)(m + 8) * VDIM_, 0, 0);
    const float4 vv = *(const float4*)(base + (size_t)m * VDIM_);
    float wm = wsh[m];
    a0 += wm * vv.x; a1 += wm * vv.y; a2 += wm * vv.z; a3 += wm * vv.w;
  }
  float4 o; o.x = a0; o.y = a1; o.z = a2; o.w = a3;
  *(float4*)(feats + (size_t)b * VDIM_ + vbase) = o;
}

// ---------------------------------------------------------------------------
// K6: x = feats @ W_lin^T + b_lin.  M=64(batch), N=1024, K=2048. bf16 WMMA.
// Double-buffered LDS tiles. Grid 8 (column blocks of 128). 8 waves.
// ---------------------------------------------------------------------------
__global__ void k_lin(const float* __restrict__ feats,
                      const float* __restrict__ W_lin,
                      const float* __restrict__ b_lin,
                      float* __restrict__ xbuf) {
  __shared__ unsigned short Alds[2][64 * 32];
  __shared__ unsigned short Blds[2][128 * 32];

  int tid = threadIdx.x;
  int lane = tid & 31, w = tid >> 5;
  int e0 = blockIdx.x * 128;

  v8f acc[4];
#pragma unroll
  for (int mt = 0; mt < 4; mt++) acc[mt] = zero8();

  int rA = tid >> 2, kA = (tid & 3) * 8;
  int rB = tid >> 1, kB = (tid & 1) * 16;
  const float* srcArow = feats + (size_t)rA * VDIM_ + kA;
  const float* srcBrow = W_lin + (size_t)(e0 + rB) * VDIM_ + kB;

  auto stage = [&](int buf, int kt) {
    int v0 = kt * 32;
    float a[8], bb[16];
#pragma unroll
    for (int j = 0; j < 8; j++) a[j] = srcArow[v0 + j];
#pragma unroll
    for (int j = 0; j < 16; j++) bb[j] = srcBrow[v0 + j];
#pragma unroll
    for (int j = 0; j < 8; j++) Alds[buf][rA * 32 + kA + j] = f2bf(a[j]);
#pragma unroll
    for (int j = 0; j < 16; j++) Blds[buf][rB * 32 + kB + j] = f2bf(bb[j]);
  };

  stage(0, 0);
  int cur = 0;
  for (int kt = 0; kt < VDIM_ / 32; kt++) {
    __syncthreads();
    if (kt + 1 < VDIM_ / 32) stage(cur ^ 1, kt + 1);
    v16bf bf = load_b_frag(Blds[cur], 16 * w, 32, 0, lane);
#pragma unroll
    for (int mt = 0; mt < 4; mt++) {
      v16bf af = load_a_frag(Alds[cur], 16 * mt, 32, 0, lane);
      acc[mt] = wmma_bf16(af, bf, acc[mt]);
    }
    cur ^= 1;
  }

  int lo = lane & 15, hi = lane >> 4;
  int e = e0 + 16 * w + lo;
  float bi = b_lin[e];
#pragma unroll
  for (int mt = 0; mt < 4; mt++) {
#pragma unroll
    for (int r = 0; r < 8; r++) {
      int row = 16 * mt + r + 8 * hi;   // batch index 0..63
      xbuf[(size_t)row * EMBED_ + e] = acc[mt][r] + bi;
    }
  }
}

// ---------------------------------------------------------------------------
// K7: BatchNorm1d (training-mode biased batch stats) over B=64.
// Grid 4 x 256: one thread per embed column.
// ---------------------------------------------------------------------------
__global__ void k_bn(const float* __restrict__ xbuf,
                     const float* __restrict__ gamma,
                     const float* __restrict__ beta,
                     float* __restrict__ out) {
  int e = blockIdx.x * 256 + threadIdx.x;
  float s = 0.f, s2 = 0.f;
#pragma unroll 4
  for (int b = 0; b < B_; b++) {
    float x = xbuf[(size_t)b * EMBED_ + e];
    s += x; s2 += x * x;
  }
  float mu = s * (1.f / B_);
  float var = s2 * (1.f / B_) - mu * mu;
  float inv = rsqrtf(var + 1e-5f);
  float g = gamma[e], be = beta[e];
#pragma unroll 4
  for (int b = 0; b < B_; b++) {
    out[(size_t)b * EMBED_ + e] =
        g * (xbuf[(size_t)b * EMBED_ + e] - mu) * inv + be;
  }
}

// ---------------------------------------------------------------------------
// launch
// ---------------------------------------------------------------------------
extern "C" void kernel_launch(void* const* d_in, const int* in_sizes, int n_in,
                              void* d_out, int out_size, void* d_ws,
                              size_t ws_size, hipStream_t stream) {
  const float* Vmat = (const float*)d_in[0];
  const float* U1_v = (const float*)d_in[1];
  const float* U1_g = (const float*)d_in[2];
  const float* U1_b = (const float*)d_in[3];
  const float* U2_v = (const float*)d_in[4];
  const float* U2_g = (const float*)d_in[5];
  const float* U2_b = (const float*)d_in[6];
  const float* W_lin = (const float*)d_in[7];
  const float* b_lin = (const float*)d_in[8];
  const float* bn_gamma = (const float*)d_in[9];
  const float* bn_beta = (const float*)d_in[10];
  float* out = (float*)d_out;

  char* ws = (char*)d_ws;
  // workspace layout (bytes)
  float* scale = (float*)(ws + 0);                     //   512 B (128 f32)
  float* dbuf  = (float*)(ws + 1024);                  //  64 KB
  float* wbuf  = (float*)(ws + 1024 + 65536);          //  64 KB
  float* feats = (float*)(ws + 1024 + 2 * 65536);      // 512 KB
  float* xbuf  = (float*)(ws + 1024 + 2 * 65536 + 524288);            // 256 KB
  unsigned short* P =
      (unsigned short*)(ws + 1024 + 2 * 65536 + 524288 + 262144);     // 4 MB

  k_scales<<<128, 256, 0, stream>>>(U1_v, U1_g, U2_v, U2_g, scale);
  k_proj<<<dim3(4, 64), 256, 0, stream>>>(Vmat, U1_v, U2_v, U1_b, U2_b, scale, P);
  k_diag<<<64, 256, 0, stream>>>(P, dbuf);
  k_colsum<<<dim3(4, 64), 256, 0, stream>>>(P, dbuf, wbuf);
  k_feats<<<dim3(2, 64), 256, 0, stream>>>(Vmat, wbuf, feats);
  k_lin<<<8, 256, 0, stream>>>(feats, W_lin, b_lin, xbuf);
  k_bn<<<4, 256, 0, stream>>>(xbuf, bn_gamma, bn_beta, out);
}
